// TransformerLayer_34591666602085
// MI455X (gfx1250) — matrix operand order repaired and checked by reading, not compiled, not run on hardware
//
#include <hip/hip_runtime.h>

typedef __bf16 bf16;
typedef __attribute__((ext_vector_type(4)))  __bf16 v4bf;
typedef __attribute__((ext_vector_type(8)))  __bf16 v8bf;
typedef __attribute__((ext_vector_type(16))) __bf16 v16bf;
typedef __attribute__((ext_vector_type(8)))  float  v8f;
typedef __attribute__((ext_vector_type(4)))  unsigned int v4u;
typedef __attribute__((ext_vector_type(8)))  int v8i;
typedef __attribute__((ext_vector_type(4)))  int v4i;

#define BATCH  4
#define SEQN   1024
#define SEQM   1024
#define DMODEL 1024
#define NHEAD  8
#define DHEAD  128
#define DFF    2048

// ---------------------------------------------------------------------------
// WMMA helpers (CDNA5 wave32: D = A(16x32) * B(32x16) + C, fp32 accum)
// ---------------------------------------------------------------------------
__device__ __forceinline__ v8f wmma_bf16(v16bf a, v16bf b, v8f c) {
  return __builtin_amdgcn_wmma_f32_16x16x32_bf16(false, a, false, b, (short)0, c,
                                                 false, false);
}

// A-matrix 16x32 bf16: lanes 0-15 hold row M=lane with
// K = k0..k0+15; lanes 16-31 hold K = k0+16..k0+31 (matches contiguous B).
__device__ __forceinline__ v16bf load_a16(const bf16* __restrict__ row, int k0, int lh) {
  v8bf lo = *(const v8bf*)(row + k0 + lh * 16);
  v8bf hi = *(const v8bf*)(row + k0 + lh * 16 + 8);
  return __builtin_shufflevector(lo, hi, 0, 1, 2, 3, 4, 5, 6, 7,
                                 8, 9, 10, 11, 12, 13, 14, 15);
}

// ---------------------------------------------------------------------------
// fp32 -> bf16 conversion (vectorized x4)
// ---------------------------------------------------------------------------
__global__ __launch_bounds__(256) void cvt_f32_bf16(const float* __restrict__ in,
                                                    bf16* __restrict__ out, int n4) {
  int i = blockIdx.x * 256 + threadIdx.x;
  if (i < n4) {
    float4 v = ((const float4*)in)[i];
    v4bf r;
    r[0] = (bf16)v.x; r[1] = (bf16)v.y; r[2] = (bf16)v.z; r[3] = (bf16)v.w;
    ((v4bf*)out)[i] = r;
  }
}

// V (B,M,D) bf16 -> Vt (B,H,C,M) bf16 so the attn@V B-operand is M-contiguous.
__global__ __launch_bounds__(256) void transpose_v(const bf16* __restrict__ v,
                                                   bf16* __restrict__ vt) {
  size_t i = (size_t)blockIdx.x * 256 + threadIdx.x;  // over B*M*D
  int d = (int)(i % DMODEL);
  int m = (int)((i / DMODEL) % SEQM);
  int b = (int)(i / ((size_t)DMODEL * SEQM));
  int h = d / DHEAD, c = d % DHEAD;
  vt[(((size_t)b * NHEAD + h) * DHEAD + c) * SEQM + m] = v[i];
}

// ---------------------------------------------------------------------------
// Generic WMMA GEMM:  out[r, c] = act( A[r,:] . W[c,:] + bias[c] + resid[r,c] )
// A: rows x K bf16 (row-major), W: cols x K bf16 (row-major, i.e. x @ W^T).
// Each wave computes a 16x64 tile (A reused across 4 B tiles).
// ---------------------------------------------------------------------------
__global__ __launch_bounds__(256) void gemm_wmma_bf16(
    const bf16* __restrict__ A, const bf16* __restrict__ W,
    const float* __restrict__ bias, const float* __restrict__ resid,
    float* __restrict__ outF, bf16* __restrict__ outB,
    int rows, int K, int cols, int doRelu) {
  int tid  = blockIdx.x * 256 + threadIdx.x;
  int wave = tid >> 5;
  int lane = tid & 31;
  int colT = cols >> 6;                       // tiles of 64 columns
  int totalWaves = (rows >> 4) * colT;
  if (wave >= totalWaves) return;             // wave-uniform: EXEC stays full
  int rt = wave / colT;
  int ct = wave % colT;
  int lm = lane & 15, lh = lane >> 4;

  v8f z = {0.f, 0.f, 0.f, 0.f, 0.f, 0.f, 0.f, 0.f};
  v8f acc[4];
#pragma unroll
  for (int j = 0; j < 4; ++j) acc[j] = z;

  const bf16* arow  = A + (size_t)(rt * 16 + lm) * K;
  const bf16* wbase = W + (size_t)(ct * 64 + lm) * K + lh * 16;

  for (int k0 = 0; k0 < K; k0 += 32) {
    __builtin_prefetch(arow + k0 + 256, 0, 0);     // global_prefetch_b8
    v16bf a = load_a16(arow, k0, lh);
#pragma unroll
    for (int j = 0; j < 4; ++j) {
      v16bf bm = *(const v16bf*)(wbase + (size_t)j * 16 * K + k0);
      acc[j] = wmma_bf16(a, bm, acc[j]);
    }
  }

  // C/D layout: VGPR r, lane l -> row = r + 8*(l>>4), col = l&15
#pragma unroll
  for (int j = 0; j < 4; ++j) {
    int col = ct * 64 + j * 16 + lm;
    float bval = bias[col];
#pragma unroll
    for (int r = 0; r < 8; ++r) {
      int row = rt * 16 + r + lh * 8;
      float v = acc[j][r] + bval;
      if (resid) v += resid[(size_t)row * cols + col];
      if (doRelu) v = fmaxf(v, 0.f);
      size_t oi = (size_t)row * cols + col;
      if (outF) outF[oi] = v;
      if (outB) outB[oi] = (bf16)v;
    }
  }
}

// ---------------------------------------------------------------------------
// Fused attention: one block per (b, h, 16-row N tile).
//   TDM stages the 16x128 Q tile into LDS (tensor_load_to_lds + tensorcnt).
//   scores (WMMA, QK^T over C=128) -> scale/factor/weight/mask -> LDS fp32
//   -> per-row softmax (float4 + shfl reductions) -> vectorized attn store
//      + bf16 probability copy in LDS
//   -> hidden = attn @ V (WMMA over K=M=1024, A-operands via ds_load) -> bf16
// ---------------------------------------------------------------------------
__global__ __launch_bounds__(256) void attn_fused(
    const bf16* __restrict__ qB, const bf16* __restrict__ kB,
    const bf16* __restrict__ vtB,
    const float* __restrict__ attf, const float* __restrict__ memw,
    const int* __restrict__ memMask, const int* __restrict__ attMask,
    float* __restrict__ attnOut, bf16* __restrict__ hiddenB) {
  __shared__ float sS[16 * SEQM];   // 64 KB fp32 scores
  __shared__ bf16  sP[16 * SEQM];   // 32 KB bf16 probabilities
  __shared__ bf16  sQ[16 * DHEAD];  //  4 KB Q tile (TDM destination)

  int bid = blockIdx.x;
  int nT = bid & 63;               // N/16 = 64
  int h  = (bid >> 6) & 7;
  int b  = bid >> 9;
  int nBase = nT * 16;
  int tid = threadIdx.x;
  int wave = tid >> 5, lane = tid & 31;
  int lm = lane & 15, lh = lane >> 4;
  const float scale = 0.08838834764831845f;  // 1/sqrt(128)

  // ---- Stage 0: TDM-stage Q tile (16 rows x 128 bf16, row stride 1024) ----
  if (wave == 0) {
    unsigned long long ga = (unsigned long long)(uintptr_t)(
        qB + ((size_t)(b * SEQN + nBase) * DMODEL + h * DHEAD));
    unsigned int ldsa = (unsigned int)(uintptr_t)&sQ[0];
    // D# group0: count=1 | lds_addr | global_addr[56:0] | type=2
    v4u g0 = {1u, ldsa, (unsigned int)ga,
              ((unsigned int)(ga >> 32) & 0x01FFFFFFu) | (2u << 30)};
    // D# group1: data_size=2B; tensor_dim0=128; tensor_dim1=16;
    //            tile_dim0=128; tile_dim1=16; tensor_dim0_stride=1024
    v8i g1 = {(int)(1u << 16),    // workgroup_mask=0, data_size=1 (2 bytes)
              (int)(128u << 16),  // tensor_dim0[15:0] in bits 63:48
              (int)(16u << 16),   // tensor_dim0 hi=0, tensor_dim1[15:0]=16
              (int)(128u << 16),  // tensor_dim1 hi=0, tile_dim0=128
              16,                 // tile_dim1=16, tile_dim2=0
              1024,               // tensor_dim0_stride[31:0]
              0, 0};              // stride hi / tensor_dim1_stride
    v4i gz4 = {0, 0, 0, 0};
    v8i gz8 = {0, 0, 0, 0, 0, 0, 0, 0};
    __builtin_amdgcn_tensor_load_to_lds(g0, g1, gz4, gz4, gz8, 0);
    __builtin_amdgcn_s_wait_tensorcnt(0);
  }
  __syncthreads();

  // ---- Stage 1: scores = (Q K^T) * scale * factors * weights, masked ----
  v8f z = {0.f, 0.f, 0.f, 0.f, 0.f, 0.f, 0.f, 0.f};
  v16bf aQ[4];
  {
    const bf16* qrow = &sQ[lm * DHEAD];
#pragma unroll
    for (int kk = 0; kk < 4; ++kk) aQ[kk] = load_a16(qrow, kk * 32, lh);
  }
  for (int ctile = wave; ctile < SEQM / 16; ctile += 8) {
    int mBase = ctile * 16;
    const bf16* krow =
        kB + ((size_t)(b * SEQM + mBase + lm) * DMODEL + h * DHEAD + lh * 16);
    v8f acc = z;
#pragma unroll
    for (int kk = 0; kk < 4; ++kk) {
      v16bf bm = *(const v16bf*)(krow + kk * 32);
      acc = wmma_bf16(aQ[kk], bm, acc);
    }
#pragma unroll
    for (int r = 0; r < 8; ++r) {
      int qr = r + lh * 8;
      int n  = nBase + qr;
      int m  = mBase + lm;
      float v = acc[r] * scale;
      v *= attf[((size_t)b * SEQN + n) * SEQM + m];
      v *= memw[b * SEQM + m];
      if (memMask[b * SEQM + m] || attMask[((size_t)b * SEQN + n) * SEQM + m])
        v = -__builtin_inff();
      sS[qr * SEQM + m] = v;
    }
  }
  __syncthreads();

  // ---- Stage 2: row softmax (float4) + attn store + bf16 prob copy ----
#pragma unroll
  for (int rr = 0; rr < 2; ++rr) {
    int row = wave * 2 + rr;
    float4* sr4 = (float4*)(sS + row * SEQM);
    float mx = -__builtin_inff();
    for (int m4 = lane; m4 < SEQM / 4; m4 += 32) {
      float4 v = sr4[m4];
      mx = fmaxf(mx, fmaxf(fmaxf(v.x, v.y), fmaxf(v.z, v.w)));
    }
#pragma unroll
    for (int off = 16; off; off >>= 1) mx = fmaxf(mx, __shfl_xor(mx, off, 32));
    float sum = 0.f;
    for (int m4 = lane; m4 < SEQM / 4; m4 += 32) {
      float4 v = sr4[m4];
      v.x = __expf(v.x - mx); v.y = __expf(v.y - mx);
      v.z = __expf(v.z - mx); v.w = __expf(v.w - mx);
      sr4[m4] = v;
      sum += v.x + v.y + v.z + v.w;
    }
#pragma unroll
    for (int off = 16; off; off >>= 1) sum += __shfl_xor(sum, off, 32);
    float inv = 1.f / sum;
    int n = nBase + row;
    float4* dst4 =
        (float4*)(attnOut + (((size_t)(b * NHEAD + h) * SEQN + n) * SEQM));
    v4bf* sp4 = (v4bf*)(sP + row * SEQM);
    for (int m4 = lane; m4 < SEQM / 4; m4 += 32) {
      float4 v = sr4[m4];
      v.x *= inv; v.y *= inv; v.z *= inv; v.w *= inv;
      dst4[m4] = v;
      v4bf pb;
      pb[0] = (bf16)v.x; pb[1] = (bf16)v.y; pb[2] = (bf16)v.z; pb[3] = (bf16)v.w;
      sp4[m4] = pb;
    }
  }
  __syncthreads();

  // ---- Stage 3: hidden = attn @ V  (each wave: one 16-col tile of C) ----
  {
    int cBase = wave * 16;
    const bf16* vrow =
        vtB + (((size_t)(b * NHEAD + h) * DHEAD + cBase + lm) * SEQM + lh * 16);
    const bf16* prow = &sP[lm * SEQM];
    v8f acc = z;
    for (int k0 = 0; k0 < SEQM; k0 += 32) {
      v16bf a  = load_a16(prow, k0, lh);
      v16bf bm = *(const v16bf*)(vrow + k0);
      acc = wmma_bf16(a, bm, acc);
    }
#pragma unroll
    for (int r = 0; r < 8; ++r) {
      int n = nBase + r + lh * 8;
      int c = cBase + lm;
      hiddenB[(size_t)(b * SEQN + n) * DMODEL + h * DHEAD + c] = (bf16)acc[r];
    }
  }
}

// ---------------------------------------------------------------------------
// LayerNorm over D=1024 per row: one 256-thread block per row, 4 elems/thread.
// ---------------------------------------------------------------------------
__global__ __launch_bounds__(256) void layernorm_k(
    const float* __restrict__ x, const float* __restrict__ g,
    const float* __restrict__ bta, float* __restrict__ outF,
    bf16* __restrict__ outB) {
  __shared__ float rsum[8], rsq[8];
  int row = blockIdx.x;
  int tid = threadIdx.x;
  const float4 xv = ((const float4*)(x + (size_t)row * DMODEL))[tid];
  float s = xv.x + xv.y + xv.z + xv.w;
  float q = xv.x * xv.x + xv.y * xv.y + xv.z * xv.z + xv.w * xv.w;
#pragma unroll
  for (int off = 16; off; off >>= 1) {
    s += __shfl_xor(s, off, 32);
    q += __shfl_xor(q, off, 32);
  }
  int wave = tid >> 5, lane = tid & 31;
  if (lane == 0) { rsum[wave] = s; rsq[wave] = q; }
  __syncthreads();
  float ts = 0.f, tq = 0.f;
#pragma unroll
  for (int i = 0; i < 8; ++i) { ts += rsum[i]; tq += rsq[i]; }
  float mean = ts * (1.f / DMODEL);
  float var  = tq * (1.f / DMODEL) - mean * mean;
  float inv  = rsqrtf(var + 1e-5f);
  float4 gv = ((const float4*)g)[tid];
  float4 bv = ((const float4*)bta)[tid];
  float4 y;
  y.x = (xv.x - mean) * inv * gv.x + bv.x;
  y.y = (xv.y - mean) * inv * gv.y + bv.y;
  y.z = (xv.z - mean) * inv * gv.z + bv.z;
  y.w = (xv.w - mean) * inv * gv.w + bv.w;
  if (outF) ((float4*)(outF + (size_t)row * DMODEL))[tid] = y;
  if (outB) {
    v4bf r;
    r[0] = (bf16)y.x; r[1] = (bf16)y.y; r[2] = (bf16)y.z; r[3] = (bf16)y.w;
    ((v4bf*)(outB + (size_t)row * DMODEL))[tid] = r;
  }
}

// ---------------------------------------------------------------------------
// Host-side orchestration
// ---------------------------------------------------------------------------
extern "C" void kernel_launch(void* const* d_in, const int* in_sizes, int n_in,
                              void* d_out, int out_size, void* d_ws, size_t ws_size,
                              hipStream_t stream) {
  const float* input_states      = (const float*)d_in[0];
  const float* memory_states     = (const float*)d_in[1];
  const float* memory_weights    = (const float*)d_in[2];
  const int*   memory_masks      = (const int*)d_in[3];
  const float* attention_factors = (const float*)d_in[4];
  const int*   attention_masks   = (const int*)d_in[5];
  const float* Wq = (const float*)d_in[6];
  const float* bq = (const float*)d_in[7];
  const float* Wk = (const float*)d_in[8];
  const float* bk = (const float*)d_in[9];
  const float* Wv = (const float*)d_in[10];
  const float* bv = (const float*)d_in[11];
  const float* Wo = (const float*)d_in[12];
  const float* bo = (const float*)d_in[13];
  const float* g1 = (const float*)d_in[14];
  const float* b1 = (const float*)d_in[15];
  const float* We = (const float*)d_in[16];
  const float* be = (const float*)d_in[17];
  const float* Ws = (const float*)d_in[18];
  const float* bs = (const float*)d_in[19];
  const float* g2 = (const float*)d_in[20];
  const float* b2 = (const float*)d_in[21];

  char* ws = (char*)d_ws;
  size_t off = 0;
  auto alloc = [&](size_t bytes) -> void* {
    void* p = ws + off;
    off += (bytes + 255) & ~(size_t)255;
    return p;
  };

  const size_t BN = (size_t)BATCH * SEQN;   // 4096
  const size_t BM = (size_t)BATCH * SEQM;   // 4096
  const size_t DD = (size_t)DMODEL * DMODEL;

  bf16* XB    = (bf16*)alloc(BN * DMODEL * 2);
  bf16* MemB  = (bf16*)alloc(BM * DMODEL * 2);
  bf16* WqB   = (bf16*)alloc(DD * 2);
  bf16* WkB   = (bf16*)alloc(DD * 2);
  bf16* WvB   = (bf16*)alloc(DD * 2);
  bf16* WoB   = (bf16*)alloc(DD * 2);
  bf16* WeB   = (bf16*)alloc((size_t)DFF * DMODEL * 2);
  bf16* WsB   = (bf16*)alloc((size_t)DMODEL * DFF * 2);
  bf16* qB    = (bf16*)alloc(BN * DMODEL * 2);
  bf16* kB    = (bf16*)alloc(BM * DMODEL * 2);
  bf16* vB    = (bf16*)alloc(BM * DMODEL * 2);
  bf16* vtB   = (bf16*)alloc(BM * DMODEL * 2);
  bf16* hidB  = (bf16*)alloc(BN * DMODEL * 2);
  float* resF = (float*)alloc(BN * DMODEL * 4);   // reused: attn resid, then ffn2
  float* out1F = (float*)alloc(BN * DMODEL * 4);
  bf16* out1B = (bf16*)alloc(BN * DMODEL * 2);
  bf16* hffB  = (bf16*)alloc(BN * DFF * 2);

  float* outMain = (float*)d_out;                       // (B,N,D)
  float* outAttn = (float*)d_out + BN * DMODEL;         // (B,H,N,M)

  // --- fp32 -> bf16 conversions ---
  auto cvt = [&](const float* src, bf16* dst, size_t n) {
    int n4 = (int)(n / 4);
    cvt_f32_bf16<<<(n4 + 255) / 256, 256, 0, stream>>>(src, dst, n4);
  };
  cvt(input_states, XB, BN * DMODEL);
  cvt(memory_states, MemB, BM * DMODEL);
  cvt(Wq, WqB, DD);
  cvt(Wk, WkB, DD);
  cvt(Wv, WvB, DD);
  cvt(Wo, WoB, DD);
  cvt(We, WeB, (size_t)DFF * DMODEL);
  cvt(Ws, WsB, (size_t)DMODEL * DFF);

  // --- GEMM launcher: 8 waves / block, 16x64 tile / wave ---
  auto gemm = [&](const bf16* A, const bf16* W, const float* bias,
                  const float* resid, float* oF, bf16* oB,
                  int rows, int K, int cols, int relu) {
    int totalWaves = (rows / 16) * (cols / 64);
    int blocks = totalWaves / 8;
    gemm_wmma_bf16<<<blocks, 256, 0, stream>>>(A, W, bias, resid, oF, oB,
                                               rows, K, cols, relu);
  };

  // QKV projections (bf16 outputs feed attention)
  gemm(XB,   WqB, bq, nullptr, nullptr, qB, (int)BN, DMODEL, DMODEL, 0);
  gemm(MemB, WkB, bk, nullptr, nullptr, kB, (int)BM, DMODEL, DMODEL, 0);
  gemm(MemB, WvB, bv, nullptr, nullptr, vB, (int)BM, DMODEL, DMODEL, 0);

  // V -> Vt (B,H,C,M)
  {
    size_t n = BM * DMODEL;
    transpose_v<<<(unsigned)(n / 256), 256, 0, stream>>>(vB, vtB);
  }

  // Fused attention: TDM Q staging -> scores -> softmax -> attn out + hidden
  attn_fused<<<BATCH * NHEAD * (SEQN / 16), 256, 0, stream>>>(
      qB, kB, vtB, attention_factors, memory_weights, memory_masks,
      attention_masks, outAttn, hidB);

  // Output projection + residual, then LN1
  gemm(hidB, WoB, bo, input_states, resF, nullptr, (int)BN, DMODEL, DMODEL, 0);
  layernorm_k<<<(unsigned)BN, 256, 0, stream>>>(resF, g1, b1, out1F, out1B);

  // FFN: expand + ReLU, squeeze + residual, LN2 -> main output
  gemm(out1B, WeB, be, nullptr, nullptr, hffB, (int)BN, DMODEL, DFF, 1);
  gemm(hffB, WsB, bs, out1F, resF, nullptr, (int)BN, DFF, DMODEL, 0);
  layernorm_k<<<(unsigned)BN, 256, 0, stream>>>(resF, g2, b2, outMain, nullptr);

  (void)in_sizes; (void)n_in; (void)out_size; (void)ws_size;
}
